// BlockShuffleLayer_5909874999579
// MI455X (gfx1250) — matrix operand order repaired and checked by reading, not compiled, not run on hardware
//
#include <hip/hip_runtime.h>
#include <stdint.h>

typedef __attribute__((ext_vector_type(16))) __bf16        v16bf;
typedef __attribute__((ext_vector_type(2)))  __bf16        v2bf;
typedef __attribute__((ext_vector_type(2)))  float         f32x2;
typedef __attribute__((ext_vector_type(8)))  float         v8f;
typedef __attribute__((ext_vector_type(4)))  float         f32x4;
typedef __attribute__((ext_vector_type(4)))  unsigned int  u32x4;

// Packed f32x2 -> bf16x2 (RNE). Prefer the single-instruction packed cvt.
__device__ __forceinline__ unsigned int pack_bf16x2(float a, float b) {
#if __has_builtin(__builtin_amdgcn_cvt_pk_bf16_f32)
    union { v2bf v; unsigned int u; } c;
    c.v = __builtin_amdgcn_cvt_pk_bf16_f32(a, b);
    return c.u;
#else
    f32x2 f = { a, b };
    union { v2bf v; unsigned int u; } c;
    c.v = __builtin_convertvector(f, v2bf);
    return c.u;
#endif
}

union Frag { v16bf v; u32x4 q[2]; unsigned short u[16]; };

// ---------------------------------------------------------------------------
// Stage 1: out1_shuf[b, l*256 + r] = stage-1 result at flat col (16r+l),
// i.e. the butterfly shuffle is folded into the store addressing.
// grid: (4096/64, 16*4), block 256 (8 waves). Per-wave: 1 M-subtile x 2 N-subtiles.
// ---------------------------------------------------------------------------
__global__ __launch_bounds__(256) void bs_stage1(
        const float* __restrict__ x, const float* __restrict__ w1,
        unsigned short* __restrict__ out1)
{
    __shared__ unsigned short lA[64][40];   // [m][p] bf16, 20-dword row stride
    __shared__ unsigned short lB[64][40];   // [q][p] bf16

    const int tid  = threadIdx.x;
    const int lane = tid & 31;
    const int wv   = tid >> 5;

    const int b0 = blockIdx.x * 64;
    const int k  = blockIdx.y >> 2;
    const int q0 = (blockIdx.y & 3) * 64;

    const int msub = wv >> 1;        // 0..3  (M subtile)
    const int nsb  = (wv & 1) * 2;   // 0 / 2 (two 16-wide N subtiles per wave)

    const float* xblk = x  + (size_t)b0 * 4096 + (size_t)k * 256;
    const float* wblk = w1 + (size_t)k * 65536 + (size_t)q0 * 256;

    v8f acc0 = {}; v8f acc1 = {};

    const int ldRow = tid >> 2;        // 0..63
    const int ldCol = (tid & 3) * 8;   // 0,8,16,24

    for (int p0 = 0; p0 < 256; p0 += 32) {
        // cooperative load + packed f32->bf16 convert into LDS
        f32x4 a0 = *(const f32x4*)(xblk + (size_t)ldRow * 4096 + p0 + ldCol);
        f32x4 a1 = *(const f32x4*)(xblk + (size_t)ldRow * 4096 + p0 + ldCol + 4);
        f32x4 w0 = *(const f32x4*)(wblk + (size_t)ldRow * 256  + p0 + ldCol);
        f32x4 w1v= *(const f32x4*)(wblk + (size_t)ldRow * 256  + p0 + ldCol + 4);
        u32x4 pa = { pack_bf16x2(a0.x, a0.y), pack_bf16x2(a0.z, a0.w),
                     pack_bf16x2(a1.x, a1.y), pack_bf16x2(a1.z, a1.w) };
        u32x4 pb = { pack_bf16x2(w0.x, w0.y), pack_bf16x2(w0.z, w0.w),
                     pack_bf16x2(w1v.x,w1v.y), pack_bf16x2(w1v.z,w1v.w) };
        *(u32x4*)(&lA[ldRow][ldCol]) = pa;
        *(u32x4*)(&lB[ldRow][ldCol]) = pb;
        __syncthreads();

        if (p0 + 32 < 256) {   // prefetch next K-step tiles (global_prefetch_b8)
            __builtin_prefetch(xblk + (size_t)ldRow * 4096 + p0 + 32 + ldCol, 0, 0);
            __builtin_prefetch(wblk + (size_t)ldRow * 256  + p0 + 32 + ldCol, 0, 0);
        }

        // A fragment: 16x32 bf16, lanes 0-15 rows M, K halves {0-7,16-23}/{8-15,24-31}
        const int mrow  = msub * 16 + (lane & 15);
        const int khalf = (lane >> 4) * 8;
        Frag a;
        a.q[0] = *(const u32x4*)(&lA[mrow][khalf]);
        a.q[1] = *(const u32x4*)(&lA[mrow][16 + khalf]);

        // B fragments: 32x16 bf16, lane col = q, K halves {0-7,16-23}/{8-15,24-31}
        const int koff = (lane >> 4) * 8;
        const int qr0  = nsb * 16 + (lane & 15);
        const int qr1  = (nsb + 1) * 16 + (lane & 15);
        Frag bA, bB;
        bA.q[0] = *(const u32x4*)(&lB[qr0][koff]);
        bA.q[1] = *(const u32x4*)(&lB[qr0][koff + 16]);
        bB.q[0] = *(const u32x4*)(&lB[qr1][koff]);
        bB.q[1] = *(const u32x4*)(&lB[qr1][koff + 16]);

        acc0 = __builtin_amdgcn_wmma_f32_16x16x32_bf16(false, a.v, false, bA.v,
                                                       (short)0, acc0, false, false);
        acc1 = __builtin_amdgcn_wmma_f32_16x16x32_bf16(false, a.v, false, bB.v,
                                                       (short)0, acc1, false, false);
        __syncthreads();
    }

    // Store into SHUFFLED layout: flat = k*256+q -> col = (flat&15)*256 + (flat>>4).
    // acc1's flat = acc0's flat + 16  ->  adjacent shuffled column: pack as one b32.
    const int flat0 = k * 256 + q0 + nsb * 16 + (lane & 15);
    const int scol  = (flat0 & 15) * 256 + (flat0 >> 4);   // even
    const int row0  = b0 + msub * 16 + (lane >> 4) * 8;
#pragma unroll
    for (int i = 0; i < 8; ++i) {
        *(unsigned int*)(out1 + (size_t)(row0 + i) * 4096 + scol)
            = pack_bf16x2(acc0[i], acc1[i]);
    }
}

// ---------------------------------------------------------------------------
// Stage 2: out[b, s*16+l] = sum_r out1_shuf[b, l*256+r] * w2[l, s, r]
// grid: (4096/32, 1024/16), block 256. Each wave owns l in {2w,2w+1}, 2 M-subtiles.
// A is contiguous per l -> b128 LDS fragments, no strided gather.
// ---------------------------------------------------------------------------
__global__ __launch_bounds__(256) void bs_stage2(
        const unsigned short* __restrict__ out1, const float* __restrict__ w2,
        float* __restrict__ out)
{
    __shared__ __align__(16) unsigned char smem[61440];
    unsigned short* lA = (unsigned short*)smem;            // [16 l][32 m][40] (40960 B)
    unsigned short* lB = (unsigned short*)(smem + 40960);  // [16 l][16 s][40] (20480 B)
    float*          lD = (float*)smem;                     // [32][260] aliases lA

    const int tid  = threadIdx.x;
    const int lane = tid & 31;
    const int wv   = tid >> 5;

    const int b0 = blockIdx.x * 32;
    const int s0 = blockIdx.y * 16;

    v8f acc[2][2] = {};   // [l-local][m-subtile]

    const int cl = tid >> 4;     // 0..15 : l for staging
    const int cm = tid & 15;     // 0..15 : m (this thread also does m+16)

    for (int r0 = 0; r0 < 256; r0 += 32) {
        // stage A: contiguous 64B runs per (l, m) from the shuffled intermediate
        const unsigned short* sA0 = out1 + (size_t)(b0 + cm) * 4096      + cl * 256 + r0;
        const unsigned short* sA1 = out1 + (size_t)(b0 + cm + 16) * 4096 + cl * 256 + r0;
        unsigned short* dA0 = lA + (cl * 32 + cm) * 40;
        unsigned short* dA1 = lA + (cl * 32 + cm + 16) * 40;
#pragma unroll
        for (int j = 0; j < 4; ++j) {
            *(u32x4*)(dA0 + j * 8) = *(const u32x4*)(sA0 + j * 8);
            *(u32x4*)(dA1 + j * 8) = *(const u32x4*)(sA1 + j * 8);
        }

        // stage B: w2[l][s0+ss][r0..r0+31], f32 -> bf16, r contiguous
        const float* srcB = w2 + ((size_t)cl * 1024 + (size_t)(s0 + cm)) * 256 + r0;
        unsigned short* dstB = lB + (cl * 16 + cm) * 40;
#pragma unroll
        for (int j = 0; j < 4; ++j) {
            f32x4 f0 = *(const f32x4*)(srcB + j * 8);
            f32x4 f1 = *(const f32x4*)(srcB + j * 8 + 4);
            u32x4 p = { pack_bf16x2(f0.x, f0.y), pack_bf16x2(f0.z, f0.w),
                        pack_bf16x2(f1.x, f1.y), pack_bf16x2(f1.z, f1.w) };
            *(u32x4*)(dstB + j * 8) = p;
        }
        __syncthreads();

        if (r0 + 32 < 256) {
            __builtin_prefetch(sA0 + 32, 0, 0);
            __builtin_prefetch(sA1 + 32, 0, 0);
            __builtin_prefetch(srcB + 32, 0, 0);
        }

        const int khalf = (lane >> 4) * 8;
        const int koff  = (lane >> 4) * 8;
        const int mlow  = lane & 15;

        Frag bfr[2], afr[2][2];
#pragma unroll
        for (int ll = 0; ll < 2; ++ll) {
            const int l = wv * 2 + ll;
            const unsigned short* brow = lB + (l * 16 + mlow) * 40;
            bfr[ll].q[0] = *(const u32x4*)(brow + koff);
            bfr[ll].q[1] = *(const u32x4*)(brow + koff + 16);
#pragma unroll
            for (int ms = 0; ms < 2; ++ms) {
                const unsigned short* arow = lA + (l * 32 + ms * 16 + mlow) * 40;
                afr[ll][ms].q[0] = *(const u32x4*)(arow + khalf);
                afr[ll][ms].q[1] = *(const u32x4*)(arow + 16 + khalf);
            }
        }
#pragma unroll
        for (int ll = 0; ll < 2; ++ll)
#pragma unroll
            for (int ms = 0; ms < 2; ++ms)
                acc[ll][ms] = __builtin_amdgcn_wmma_f32_16x16x32_bf16(
                    false, afr[ll][ms].v, false, bfr[ll].v, (short)0,
                    acc[ll][ms], false, false);
        __syncthreads();
    }

    // restage D through LDS so global stores are dense b128 rows
#pragma unroll
    for (int ll = 0; ll < 2; ++ll) {
        const int l = wv * 2 + ll;
        const int c = (lane & 15) * 16 + l;     // local col = ss*16 + l
#pragma unroll
        for (int ms = 0; ms < 2; ++ms) {
            const int r = ms * 16 + (lane >> 4) * 8;
#pragma unroll
            for (int i = 0; i < 8; ++i)
                lD[(r + i) * 260 + c] = acc[ll][ms][i];
        }
    }
    __syncthreads();

    const int orow = tid >> 3;             // 0..31
    const int ocol = (tid & 7) * 32;       // 0..224
    float* dst = out + (size_t)(b0 + orow) * 16384 + (size_t)s0 * 16 + ocol;
    const float* srcD = lD + orow * 260 + ocol;
#pragma unroll
    for (int j = 0; j < 8; ++j)
        *(f32x4*)(dst + j * 4) = *(const f32x4*)(srcD + j * 4);
}

// ---------------------------------------------------------------------------
extern "C" void kernel_launch(void* const* d_in, const int* in_sizes, int n_in,
                              void* d_out, int out_size, void* d_ws, size_t ws_size,
                              hipStream_t stream) {
    (void)in_sizes; (void)n_in; (void)out_size; (void)ws_size;
    const float* x  = (const float*)d_in[0];
    const float* w1 = (const float*)d_in[1];
    const float* w2 = (const float*)d_in[2];
    float* out = (float*)d_out;
    unsigned short* out1 = (unsigned short*)d_ws;   // 4096*4096 bf16 = 32 MB scratch

    bs_stage1<<<dim3(64, 64),  256, 0, stream>>>(x, w1, out1);
    bs_stage2<<<dim3(128, 64), 256, 0, stream>>>(out1, w2, out);
}
